// NSNet_27144193311190
// MI455X (gfx1250) — matrix operand order repaired and checked by reading, not compiled, not run on hardware
//
#include <hip/hip_runtime.h>
#include <math.h>

// ---------------- graph constants (match reference) ----------------
#define E_EDGES   300000
#define NUM_LITN  50000
#define NUM_CLSN  100000
#define BGRAPH    4
#define LITS_PER_G 12500     // 2*V
#define HDIM      64
#define NTILES    (E_EDGES/16)   // 18750
#define INV_DEMON 0.125f         // 1/sqrt(64)

typedef _Float16 half_t;
typedef __attribute__((ext_vector_type(16))) _Float16 v16h;
typedef __attribute__((ext_vector_type(8)))  float    v8f;

#define LDS_FENCE() __builtin_amdgcn_fence(__ATOMIC_ACQ_REL, "workgroup")

union Frag16 { v16h h; unsigned u[8]; };

// ---------------------------------------------------------------
// One 16(edges) x 64(out) GEMM step on a wave using WMMA f32_16x16x32_f16.
//   actIn : LDS f16 activations, row-major [16][sIn]
//   wt    : LDS f16 weights, TRANSPOSED [n][k], row stride = wstride
//   out   : either back to LDS tile (f16, bias+opt ReLU) or global f32 rows
// A-frag layout (16-bit A 16x32): lane<16 -> K 0..7 (v0..3), 16..23 (v4..7);
//                                 lane>=16 -> +8 within each half.
// B-frag layout (32x16): lane&15 = N; lanes<16 hold K 0..7,16..23 (2/VGPR), lanes>=16 K 8..15,24..31 (same order as A).
// C layout: lane&15 = N; VGPR r = M = r + 8*(lane>=16).
// ---------------------------------------------------------------
template<int KSTEPS, bool RELU, bool TO_LDS>
__device__ __forceinline__ void wmma_gemm_tile(
    const half_t* actIn, int sIn,
    half_t* actOut, int sOut,
    const half_t* wt, int wstride,
    const float* bias,
    float* gOut, int lane)
{
  const int mA   = lane & 15;
  const int kh8  = (lane >> 4) << 3;
  const int kh16 = (lane >> 4) << 4;
  Frag16 A[KSTEPS];
#pragma unroll
  for (int ks = 0; ks < KSTEPS; ++ks) {
#pragma unroll
    for (int v = 0; v < 8; ++v) {
      const int k = ks*32 + ((v>>2)<<4) + kh8 + ((v&3)<<1);
      A[ks].u[v] = *(const unsigned*)(actIn + mA*sIn + k);
    }
  }
  const int nl = lane & 15;
  const int mb = (lane >> 4) << 3;
#pragma unroll
  for (int nt = 0; nt < 4; ++nt) {
    const int ncol = (nt << 4) + nl;
    v8f C = {};
#pragma unroll
    for (int ks = 0; ks < KSTEPS; ++ks) {
      Frag16 B;
#pragma unroll
      for (int v = 0; v < 8; ++v) {
        const int k = ks*32 + ((v>>2)<<4) + kh8 + ((v&3)<<1);
        B.u[v] = *(const unsigned*)(wt + ncol*wstride + k);
      }
      C = __builtin_amdgcn_wmma_f32_16x16x32_f16(false, A[ks].h, false, B.h,
                                                 (short)0, C, false, false);
    }
    const float bv = bias[ncol];
#pragma unroll
    for (int r = 0; r < 8; ++r) {
      float val = C[r] + bv;
      if (RELU) val = fmaxf(val, 0.f);
      if (TO_LDS) actOut[(mb + r)*sOut + ncol] = (half_t)val;
      else        gOut[(mb + r)*HDIM + ncol]   = val;
    }
  }
}

// ---------------- init: h = f0*W[0] + f1*W[1] + b  (E x 2 @ 2 x 64) ----------------
__global__ __launch_bounds__(256) void init_edges(
    const float* __restrict__ feat, const float* __restrict__ W,
    const float* __restrict__ b, float* __restrict__ h)
{
  int i = blockIdx.x * blockDim.x + threadIdx.x;        // one thread per (e, k4)
  if (i >= E_EDGES * 16) return;
  int e = i >> 4, k4 = (i & 15) << 2;
  float f0 = feat[e*2], f1 = feat[e*2 + 1];
#pragma unroll
  for (int j = 0; j < 4; ++j)
    h[e*HDIM + k4 + j] = f0 * W[k4 + j] + f1 * W[64 + k4 + j] + b[k4 + j];
}

__global__ __launch_bounds__(256) void zero_f32(float* __restrict__ p, int n)
{
  for (int i = blockIdx.x * blockDim.x + threadIdx.x; i < n; i += gridDim.x * blockDim.x)
    p[i] = 0.f;
}

// segment_sum over unsorted lit_idx via fp32 atomics
__global__ __launch_bounds__(256) void scatter_add_lit(
    const float* __restrict__ src, const int* __restrict__ lit_idx,
    float* __restrict__ dst, int n)
{
  for (int i = blockIdx.x * blockDim.x + threadIdx.x; i < n; i += gridDim.x * blockDim.x) {
    int e = i >> 6, k = i & 63;
    atomicAdd(dst + lit_idx[e]*HDIM + k, src[i]);
  }
}

// clause aggregation: edges of clause c are contiguous (3c..3c+2)
__global__ __launch_bounds__(256) void agg_clause(
    const float* __restrict__ h_l2c, float* __restrict__ agg_c)
{
  int i = blockIdx.x * blockDim.x + threadIdx.x;        // one thread per (c, k4)
  if (i >= NUM_CLSN * 16) return;
  int c = i >> 4, k4 = (i & 15) << 2;
  const float* p = h_l2c + (size_t)(3*c)*HDIM + k4;
#pragma unroll
  for (int j = 0; j < 4; ++j)
    agg_c[c*HDIM + k4 + j] = p[j] + p[HDIM + j] + p[2*HDIM + j];
}

// ---------------- Phase A: literal-side fused 6-GEMM chain ----------------
//   x  = (agg_l[lit] - h_c2l)/8            -> msgl MLP3 (relu, relu, lin) -> m
//   x2 = relu([m, h_l2c] @ W0 + b0) ; relu(@Wr0) ; (@Wr1) -> new h_l2c
#define SA 136   // activation row stride in halfs (128 cols + 8 pad)

__global__ __launch_bounds__(256) void phase_a(
    const float* __restrict__ agg_l, const int* __restrict__ lit_idx,
    const float* __restrict__ h_c2l, float* __restrict__ h_l2c,
    const float* __restrict__ msgl_W, const float* __restrict__ msgl_b,
    const float* __restrict__ merge_W0, const float* __restrict__ merge_b0,
    const float* __restrict__ merge_Wr, const float* __restrict__ merge_br)
{
  __shared__ __align__(16) half_t wt[28672];       // 3*64*64 msgl | 64*128 W0 | 2*64*64 Wr (all [n][k])
  __shared__ float bias[384];                      // 3*64 | 64 | 2*64
  __shared__ __align__(16) half_t act[8][16*SA];   // per-wave activation tiles

  for (int i = threadIdx.x; i < 28672; i += 256) { // f32 -> f16, transpose to [n][k]
    float v; int dst;
    if (i < 12288)      { int l=i>>12, j=i&4095, k=j>>6, n=j&63;
                          v = msgl_W[i];   dst = (l<<12) + (n<<6) + k; }
    else if (i < 20480) { int j=i-12288, k=j>>6, n=j&63;
                          v = merge_W0[j]; dst = 12288 + (n<<7) + k; }
    else                { int j=i-20480, l=j>>12, jj=j&4095, k=jj>>6, n=jj&63;
                          v = merge_Wr[j]; dst = 20480 + (l<<12) + (n<<6) + k; }
    wt[dst] = (half_t)v;
  }
  for (int i = threadIdx.x; i < 384; i += 256)
    bias[i] = (i < 192) ? msgl_b[i] : (i < 256 ? merge_b0[i-192] : merge_br[i-256]);
  __syncthreads();

  const int lane = threadIdx.x & 31;
  const int wave = threadIdx.x >> 5;
  half_t* a = act[wave];
  const int wstride = gridDim.x << 3;
  for (int t = (blockIdx.x << 3) + wave; t < NTILES; t += wstride) {
    const int e0 = t << 4;
    const int tn = t + wstride;                        // prefetch next tile's rows
    if (tn < NTILES && lane < 16) {
      __builtin_prefetch(h_c2l + (size_t)((tn<<4) + lane)*HDIM, 0, 3);
      __builtin_prefetch(h_l2c + (size_t)((tn<<4) + lane)*HDIM, 0, 3);
    }
    // stage: cols[0..64) = (agg_l[lit]-h_c2l)*0.125 (f16); cols[64..128) = h_l2c (f16)
    for (int i = lane; i < 256; i += 32) {
      const int m = i >> 4, k4 = (i & 15) << 2;
      const int e = e0 + m;
      const int li = lit_idx[e];
      const float* pc = h_c2l + (size_t)e*HDIM + k4;
      const float* pa = agg_l + (size_t)li*HDIM + k4;
      const float* pl = h_l2c + (size_t)e*HDIM + k4;
      union { half_t h[4]; uint2 u; } q0, q1;
#pragma unroll
      for (int j = 0; j < 4; ++j) {
        q0.h[j] = (half_t)((pa[j] - pc[j]) * INV_DEMON);
        q1.h[j] = (half_t)pl[j];
      }
      *(uint2*)(a + m*SA + k4)      = q0.u;
      *(uint2*)(a + m*SA + 64 + k4) = q1.u;
    }
    LDS_FENCE();
    wmma_gemm_tile<2,true ,true >(a,SA, a,SA, wt,        64, bias,     nullptr, lane); LDS_FENCE();
    wmma_gemm_tile<2,true ,true >(a,SA, a,SA, wt+4096,   64, bias+64,  nullptr, lane); LDS_FENCE();
    wmma_gemm_tile<2,false,true >(a,SA, a,SA, wt+8192,   64, bias+128, nullptr, lane); LDS_FENCE();
    wmma_gemm_tile<4,true ,true >(a,SA, a,SA, wt+12288, 128, bias+192, nullptr, lane); LDS_FENCE();
    wmma_gemm_tile<2,true ,true >(a,SA, a,SA, wt+20480,  64, bias+256, nullptr, lane); LDS_FENCE();
    wmma_gemm_tile<2,false,false>(a,SA, nullptr,0, wt+24576, 64, bias+320,
                                  h_l2c + (size_t)e0*HDIM, lane);
  }
}

// ---------------- Phase B: clause-side fused MLP3 ----------------
#define SB 72

__global__ __launch_bounds__(256) void phase_b(
    const float* __restrict__ agg_c, const float* __restrict__ h_l2c,
    float* __restrict__ h_c2l,
    const float* __restrict__ msgc_W, const float* __restrict__ msgc_b)
{
  __shared__ __align__(16) half_t wt[12288];
  __shared__ float bias[192];
  __shared__ __align__(16) half_t act[8][16*SB];

  for (int i = threadIdx.x; i < 12288; i += 256) {
    int l=i>>12, j=i&4095, k=j>>6, n=j&63;
    wt[(l<<12) + (n<<6) + k] = (half_t)msgc_W[i];
  }
  for (int i = threadIdx.x; i < 192; i += 256) bias[i] = msgc_b[i];
  __syncthreads();

  const int lane = threadIdx.x & 31;
  const int wave = threadIdx.x >> 5;
  half_t* a = act[wave];
  const int wstride = gridDim.x << 3;
  for (int t = (blockIdx.x << 3) + wave; t < NTILES; t += wstride) {
    const int e0 = t << 4;
    const int tn = t + wstride;
    if (tn < NTILES && lane < 16)
      __builtin_prefetch(h_l2c + (size_t)((tn<<4) + lane)*HDIM, 0, 3);
    for (int i = lane; i < 256; i += 32) {
      const int m = i >> 4, k4 = (i & 15) << 2;
      const int e = e0 + m;
      const float* pg = agg_c + (size_t)(e/3)*HDIM + k4;
      const float* pl = h_l2c + (size_t)e*HDIM + k4;
      union { half_t h[4]; uint2 u; } q;
#pragma unroll
      for (int j = 0; j < 4; ++j)
        q.h[j] = (half_t)((pg[j] - pl[j]) * INV_DEMON);
      *(uint2*)(a + m*SB + k4) = q.u;
    }
    LDS_FENCE();
    wmma_gemm_tile<2,true ,true >(a,SB, a,SB, wt,      64, bias,     nullptr, lane); LDS_FENCE();
    wmma_gemm_tile<2,true ,true >(a,SB, a,SB, wt+4096, 64, bias+64,  nullptr, lane); LDS_FENCE();
    wmma_gemm_tile<2,false,false>(a,SB, nullptr,0, wt+8192, 64, bias+128,
                                  h_c2l + (size_t)e0*HDIM, lane);
  }
}

// ---------------- readout ----------------
__global__ __launch_bounds__(256) void graph_mean(
    const float* __restrict__ lemb, float* __restrict__ gmean)
{
  const int b = blockIdx.x;
  const int col = threadIdx.x & 63, part = threadIdx.x >> 6;   // 4 partials per col
  float s = 0.f;
  for (int r = part; r < LITS_PER_G; r += 4)
    s += lemb[((size_t)b*LITS_PER_G + r)*HDIM + col];
  __shared__ float red[4][64];
  red[part][col] = s;
  __syncthreads();
  if (threadIdx.x < 64)
    gmean[b*HDIM + threadIdx.x] =
      (red[0][threadIdx.x] + red[1][threadIdx.x] + red[2][threadIdx.x] + red[3][threadIdx.x])
      * (1.f / (float)LITS_PER_G);
}

__global__ __launch_bounds__(256) void readout(
    const float* __restrict__ gmean, const float* __restrict__ roW,
    const float* __restrict__ rob, const float* __restrict__ roW2,
    const float* __restrict__ rob2, float* __restrict__ out)
{
  __shared__ float x1[BGRAPH][64], x2[BGRAPH][64];
  const int b = threadIdx.x >> 6, j = threadIdx.x & 63;
  float acc = rob[j];
  for (int k = 0; k < 64; ++k) acc += gmean[b*64 + k] * roW[k*64 + j];
  x1[b][j] = fmaxf(acc, 0.f);
  __syncthreads();
  acc = rob[64 + j];
  for (int k = 0; k < 64; ++k) acc += x1[b][k] * roW[4096 + k*64 + j];
  x2[b][j] = fmaxf(acc, 0.f);
  __syncthreads();
  if (j == 0) {
    float s = rob2[0];
    for (int k = 0; k < 64; ++k) s += x2[b][k] * roW2[k];
    out[b] = 1.f / (1.f + expf(-s));
  }
}

// ---------------- host orchestration ----------------
extern "C" void kernel_launch(void* const* d_in, const int* in_sizes, int n_in,
                              void* d_out, int out_size, void* d_ws, size_t ws_size,
                              hipStream_t stream)
{
  const float* feat_l2c = (const float*)d_in[0];
  const float* feat_c2l = (const float*)d_in[1];
  const int*   lit_idx  = (const int*)d_in[2];
  // d_in[3] (cls_idx) implicit: edges sorted by clause, 3 per clause
  const float* W_l2c = (const float*)d_in[4];
  const float* b_l2c = (const float*)d_in[5];
  const float* W_c2l = (const float*)d_in[6];
  const float* b_c2l = (const float*)d_in[7];
  const float* msgl_W = (const float*)d_in[8];
  const float* msgl_b = (const float*)d_in[9];
  const float* merge_W0 = (const float*)d_in[10];
  const float* merge_b0 = (const float*)d_in[11];
  const float* merge_Wr = (const float*)d_in[12];
  const float* merge_br = (const float*)d_in[13];
  const float* msgc_W = (const float*)d_in[14];
  const float* msgc_b = (const float*)d_in[15];
  const float* ro_W  = (const float*)d_in[16];
  const float* ro_b  = (const float*)d_in[17];
  const float* ro_W2 = (const float*)d_in[18];
  const float* ro_b2 = (const float*)d_in[19];

  float* h_l2c = (float*)d_ws;                               // E*64 f32
  float* h_c2l = h_l2c + (size_t)E_EDGES*HDIM;               // E*64 f32
  float* agg_l = h_c2l + (size_t)E_EDGES*HDIM;               // NUM_LIT*64
  float* agg_c = agg_l + (size_t)NUM_LITN*HDIM;              // NUM_CLS*64
  float* gmean = agg_c + (size_t)NUM_CLSN*HDIM;              // 4*64

  const int initBlocks = (E_EDGES*16 + 255) / 256;
  init_edges<<<initBlocks, 256, 0, stream>>>(feat_l2c, W_l2c, b_l2c, h_l2c);
  init_edges<<<initBlocks, 256, 0, stream>>>(feat_c2l, W_c2l, b_c2l, h_c2l);

  for (int r = 0; r < 3; ++r) {
    zero_f32<<<2048, 256, 0, stream>>>(agg_l, NUM_LITN*HDIM);
    scatter_add_lit<<<4096, 256, 0, stream>>>(h_c2l, lit_idx, agg_l, E_EDGES*HDIM);
    phase_a<<<960, 256, 0, stream>>>(agg_l, lit_idx, h_c2l, h_l2c,
                                     msgl_W, msgl_b, merge_W0, merge_b0, merge_Wr, merge_br);
    agg_clause<<<(NUM_CLSN*16 + 255)/256, 256, 0, stream>>>(h_l2c, agg_c);
    phase_b<<<960, 256, 0, stream>>>(agg_c, h_l2c, h_c2l, msgc_W, msgc_b);
  }

  zero_f32<<<2048, 256, 0, stream>>>(agg_l, NUM_LITN*HDIM);
  scatter_add_lit<<<4096, 256, 0, stream>>>(h_c2l, lit_idx, agg_l, E_EDGES*HDIM);
  graph_mean<<<BGRAPH, 256, 0, stream>>>(agg_l, gmean);
  readout<<<1, 256, 0, stream>>>(gmean, ro_W, ro_b, ro_W2, ro_b2, (float*)d_out);
}
